// Causal_Attention_13082470383895
// MI455X (gfx1250) — matrix operand order repaired and checked by reading, not compiled, not run on hardware
//
#include <hip/hip_runtime.h>
#include <hip/hip_bf16.h>
#include <stdint.h>

// ---------------------------------------------------------------------------
// CDNA5 (gfx1250) causal attention, bf16 WMMA everywhere.
// ---------------------------------------------------------------------------

typedef __attribute__((ext_vector_type(16))) __bf16        v16bf;
typedef __attribute__((ext_vector_type(8)))  float         v8f;
typedef __attribute__((ext_vector_type(4)))  unsigned int  u32x4;
typedef __attribute__((ext_vector_type(2)))  unsigned int  u32x2;
typedef __attribute__((ext_vector_type(4)))  float         f32x4;

union V16BF { u32x4 u[2]; v16bf v; };

__device__ __forceinline__ unsigned short f2bf(float f) {
  unsigned u = __float_as_uint(f);
  return (unsigned short)((u + 0x7FFFu + ((u >> 16) & 1u)) >> 16); // RNE
}

__device__ __forceinline__ v8f wmma_bf16(v16bf a, v16bf b, v8f c) {
  // D = A(16x32 bf16) * B(32x16 bf16) + C(16x16 f32)
  return __builtin_amdgcn_wmma_f32_16x16x32_bf16(false, a, false, b,
                                                 (short)0, c, false, false);
}

// ---------------------------------------------------------------------------
// Generic tiled GEMM: C[M,N] = A[M,K] * B[K,N] * scale
//   MODE 0: A=f32, C=f32 row-major              (qk projection)
//   MODE 1: A=f32, C=bf16 scattered to vT[b,h,d,l]  (v projection, transposed)
//   MODE 2: A=bf16, C=f32 row-major             (output projection)
// Block = 128 threads (4 waves), block tile 64x64, K-step 32,
// per-wave 32x32 (2x2 WMMA fragments).
// ---------------------------------------------------------------------------
template <int MODE>
__global__ __launch_bounds__(128)
void gemm_bf16(const void* __restrict__ Ap, const float* __restrict__ B,
               void* __restrict__ Cp, int M, int N, int K, float scale) {
  __shared__ unsigned short sA[64 * 32];  // [m][k] bf16
  __shared__ unsigned short sB[64 * 32];  // [n][k] bf16 (transposed stage)

  const int tid = threadIdx.x;
  const int lane = tid & 31, wave = tid >> 5;
  const int m0 = blockIdx.y * 64, n0 = blockIdx.x * 64;
  const int wm = (wave & 1) * 32, wn = (wave >> 1) * 32;
  const int lhi = lane >> 4, llo = lane & 15;

  const v8f z = {0.f, 0.f, 0.f, 0.f, 0.f, 0.f, 0.f, 0.f};
  v8f acc[2][2];
  acc[0][0] = z; acc[0][1] = z; acc[1][0] = z; acc[1][1] = z;

  for (int k0 = 0; k0 < K; k0 += 32) {
    // ---- stage A tile (64 rows x 32 K) ----
#pragma unroll
    for (int it = 0; it < 4; ++it) {
      int idx = tid + it * 128;           // 512 quads of 4 elements
      int m = idx >> 3, kq = idx & 7;
      if (MODE == 2) {
        const unsigned short* A16 = (const unsigned short*)Ap;
        u32x2 r = *(const u32x2*)&A16[(size_t)(m0 + m) * K + k0 + kq * 4];
        *(u32x2*)&sA[m * 32 + kq * 4] = r;
      } else {
        const float* A32 = (const float*)Ap;
        f32x4 f = *(const f32x4*)&A32[(size_t)(m0 + m) * K + k0 + kq * 4];
        u32x2 r;
        r.x = (unsigned)f2bf(f.x) | ((unsigned)f2bf(f.y) << 16);
        r.y = (unsigned)f2bf(f.z) | ((unsigned)f2bf(f.w) << 16);
        *(u32x2*)&sA[m * 32 + kq * 4] = r;
      }
    }
    // ---- stage B tile (32 K x 64 N), transposed into [n][k] ----
#pragma unroll
    for (int it = 0; it < 4; ++it) {
      int idx = tid + it * 128;
      int kr = idx >> 4, nq = idx & 15;
      f32x4 f = *(const f32x4*)&B[(size_t)(k0 + kr) * N + n0 + nq * 4];
      sB[(nq * 4 + 0) * 32 + kr] = f2bf(f.x);
      sB[(nq * 4 + 1) * 32 + kr] = f2bf(f.y);
      sB[(nq * 4 + 2) * 32 + kr] = f2bf(f.z);
      sB[(nq * 4 + 3) * 32 + kr] = f2bf(f.w);
    }
    __syncthreads();

    // ---- load fragments (ISA 16-bit A/B layouts) ----
    v16bf af[2], bfm[2];
    {
      int kb = lhi * 8;
#pragma unroll
      for (int s = 0; s < 2; ++s) {
        int row = wm + s * 16 + llo;
        V16BF t;
        t.u[0] = *(const u32x4*)&sA[row * 32 + kb];
        t.u[1] = *(const u32x4*)&sA[row * 32 + 16 + kb];
        af[s] = t.v;
      }
      int kb2 = lhi * 8;
#pragma unroll
      for (int s = 0; s < 2; ++s) {
        int col = wn + s * 16 + llo;
        V16BF t;
        t.u[0] = *(const u32x4*)&sB[col * 32 + kb2];
        t.u[1] = *(const u32x4*)&sB[col * 32 + kb2 + 16];
        bfm[s] = t.v;
      }
    }
#pragma unroll
    for (int sa = 0; sa < 2; ++sa)
#pragma unroll
      for (int sb = 0; sb < 2; ++sb)
        acc[sa][sb] = wmma_bf16(af[sa], bfm[sb], acc[sa][sb]);
    __syncthreads();
  }

  // ---- epilogue ----
#pragma unroll
  for (int sa = 0; sa < 2; ++sa) {
#pragma unroll
    for (int sb = 0; sb < 2; ++sb) {
      int gmBase = m0 + wm + sa * 16 + lhi * 8;
      int gn = n0 + wn + sb * 16 + llo;
      if (MODE == 1) {
        // scatter bf16 to vT[b, h, d, l], packing consecutive-l pairs
        unsigned short* vT = (unsigned short*)Cp;
        int h = gn >> 6, d = gn & 63;
#pragma unroll
        for (int i = 0; i < 8; i += 2) {
          int gm = gmBase + i;
          int b = gm >> 11, lp = gm & 2047;
          unsigned pk = (unsigned)f2bf(acc[sa][sb][i] * scale) |
                        ((unsigned)f2bf(acc[sa][sb][i + 1] * scale) << 16);
          size_t o = ((size_t)((b * 16 + h) * 64 + d)) * 2048 + lp;
          *(unsigned*)&vT[o] = pk;
        }
      } else {
        float* C = (float*)Cp;
#pragma unroll
        for (int i = 0; i < 8; ++i)
          C[(size_t)(gmBase + i) * N + gn] = acc[sa][sb][i] * scale;
      }
    }
  }
}

// ---------------------------------------------------------------------------
// LayerNorm over each (token, head, q|k) 64-vector of qk_raw[4096, 2048].
// One wave per vector (2 elements / lane), bf16 out in [b,h,l,64].
// ---------------------------------------------------------------------------
__global__ __launch_bounds__(256)
void ln_qk(const float* __restrict__ qk, const float* __restrict__ qs,
           const float* __restrict__ qbv, const float* __restrict__ ks,
           const float* __restrict__ kbv, unsigned short* __restrict__ qo,
           unsigned short* __restrict__ ko) {
  int gt = blockIdx.x * 256 + threadIdx.x;
  int gw = gt >> 5, lane = gt & 31;
  int token = gw >> 5, rem = gw & 31;
  int h = rem >> 1, isK = rem & 1;
  const float* src = qk + (size_t)token * 2048 + h * 128 + isK * 64;
  float x0 = src[lane], x1 = src[lane + 32];
  float s = x0 + x1, sq = x0 * x0 + x1 * x1;
#pragma unroll
  for (int m = 16; m >= 1; m >>= 1) {
    s  += __shfl_xor(s, m, 32);
    sq += __shfl_xor(sq, m, 32);
  }
  float mean = s * (1.0f / 64.0f);
  float var = sq * (1.0f / 64.0f) - mean * mean;
  float rs = rsqrtf(var + 1e-6f);
  const float* sc = isK ? ks : qs;
  const float* bi = isK ? kbv : qbv;
  float y0 = (x0 - mean) * rs * sc[lane] + bi[lane];
  float y1 = (x1 - mean) * rs * sc[lane + 32] + bi[lane + 32];
  int b = token >> 11, lp = token & 2047;
  unsigned short* dst =
      (isK ? ko : qo) + ((size_t)((b * 16 + h) * 2048 + lp)) * 64;
  dst[lane] = f2bf(y0);
  dst[lane + 32] = f2bf(y1);
}

// ---------------------------------------------------------------------------
// Causal attention, flash style (streamed denominator; exp without max,
// faithful to the reference). One wave = 16 q rows; 32-key chunks.
// Q,K in [b,h,l,64] bf16; V in [b,h,d,l] bf16; out bf16 [token, 1024].
// ---------------------------------------------------------------------------
__global__ __launch_bounds__(128)
void attn_kernel(const unsigned short* __restrict__ Q,
                 const unsigned short* __restrict__ Kmat,
                 const unsigned short* __restrict__ Vt,
                 unsigned short* __restrict__ O) {
  __shared__ unsigned short sP[4][16 * 32];  // per-wave exp(S) staging
  const int tid = threadIdx.x, lane = tid & 31, wave = tid >> 5;
  const int bh = blockIdx.y;
  const int b = bh >> 4, h = bh & 15;
  const int qr0 = blockIdx.x * 64 + wave * 16;
  const unsigned short* qb = Q + (size_t)bh * 2048 * 64;
  const unsigned short* kb = Kmat + (size_t)bh * 2048 * 64;
  const unsigned short* vb = Vt + (size_t)bh * 64 * 2048;
  unsigned short* myP = sP[wave];
  const int lhi = lane >> 4, llo = lane & 15;

  // q fragments for this wave's 16 rows (d = 0..31, 32..63)
  v16bf aq[2];
  {
    int row = qr0 + llo, kbo = lhi * 8;
#pragma unroll
    for (int dc = 0; dc < 2; ++dc) {
      const unsigned short* p = qb + (size_t)row * 64 + dc * 32;
      V16BF t;
      t.u[0] = *(const u32x4*)&p[kbo];
      t.u[1] = *(const u32x4*)&p[16 + kbo];
      aq[dc] = t.v;
    }
  }

  const v8f z = {0.f, 0.f, 0.f, 0.f, 0.f, 0.f, 0.f, 0.f};
  v8f o0 = z, o1 = z, o2 = z, o3 = z;
  float dsum[8] = {0, 0, 0, 0, 0, 0, 0, 0};
  const int nCh = qr0 / 32 + 1;  // causal truncation

  for (int ch = 0; ch < nCh; ++ch) {
    int kc = ch * 32;
    // scores: S(16q x 32keys) via 4 WMMAs (2 key tiles x 2 d chunks)
    v8f sc0 = z, sc1 = z;
#pragma unroll
    for (int t = 0; t < 2; ++t) {
      int key = kc + t * 16 + llo;
#pragma unroll
      for (int dc = 0; dc < 2; ++dc) {
        const unsigned short* p = kb + (size_t)key * 64 + dc * 32 + lhi * 8;
        V16BF bt;
        bt.u[0] = *(const u32x4*)&p[0];
        bt.u[1] = *(const u32x4*)&p[16];
        if (t == 0) sc0 = wmma_bf16(aq[dc], bt.v, sc0);
        else        sc1 = wmma_bf16(aq[dc], bt.v, sc1);
      }
    }
    // masked exp(S/64); accumulate denominator; stage P in LDS as bf16
    int qm = qr0 + lhi * 8;
#pragma unroll
    for (int t = 0; t < 2; ++t) {
      int kn = kc + t * 16 + llo;
#pragma unroll
      for (int i = 0; i < 8; ++i) {
        float sv = (t == 0) ? sc0[i] : sc1[i];
        float e = (qm + i >= kn) ? __expf(sv * 0.015625f) : 0.0f;
        dsum[i] += e;
        myP[(lhi * 8 + i) * 32 + t * 16 + llo] = f2bf(e);
      }
    }
    asm volatile("s_wait_dscnt 0" ::: "memory");  // LDS store->load, cross-lane
    // reload P as A fragment
    v16bf ap;
    {
      V16BF t;
      int kbo = lhi * 8;
      t.u[0] = *(const u32x4*)&myP[llo * 32 + kbo];
      t.u[1] = *(const u32x4*)&myP[llo * 32 + 16 + kbo];
      ap = t.v;
    }
    // P(16x32) @ V(32keys x 64d): 4 WMMAs, V fragments contiguous thanks to vT
#pragma unroll
    for (int n = 0; n < 4; ++n) {
      int d = n * 16 + llo;
      const unsigned short* p = vb + (size_t)d * 2048 + kc + lhi * 8;
      V16BF bt;
      bt.u[0] = *(const u32x4*)&p[0];
      bt.u[1] = *(const u32x4*)&p[16];
      v8f& o = (n == 0) ? o0 : (n == 1) ? o1 : (n == 2) ? o2 : o3;
      o = wmma_bf16(ap, bt.v, o);
    }
  }

  // row-sum reduction across the 16-lane half-groups (matches C layout)
#pragma unroll
  for (int m = 8; m >= 1; m >>= 1)
#pragma unroll
    for (int i = 0; i < 8; ++i)
      dsum[i] += __shfl_xor(dsum[i], m, 32);
#pragma unroll
  for (int i = 0; i < 8; ++i) dsum[i] = 1.0f / dsum[i];

  // normalize + store bf16 to [token, 1024] for the output projection
  size_t rowBase = (size_t)(b * 2048 + qr0 + lhi * 8);
#pragma unroll
  for (int n = 0; n < 4; ++n) {
    v8f& o = (n == 0) ? o0 : (n == 1) ? o1 : (n == 2) ? o2 : o3;
#pragma unroll
    for (int i = 0; i < 8; ++i)
      O[(rowBase + i) * 1024 + h * 64 + n * 16 + llo] = f2bf(o[i] * dsum[i]);
  }
}

// ---------------------------------------------------------------------------
extern "C" void kernel_launch(void* const* d_in, const int* in_sizes, int n_in,
                              void* d_out, int out_size, void* d_ws,
                              size_t ws_size, hipStream_t stream) {
  (void)in_sizes; (void)n_in; (void)out_size; (void)ws_size;
  const float* X     = (const float*)d_in[0];  // [2,2048,1024]
  const float* W_qk  = (const float*)d_in[1];  // [1024,2048]
  const float* W_v   = (const float*)d_in[2];  // [1024,1024]
  const float* W_out = (const float*)d_in[3];  // [1024,1024]
  const float* q_s   = (const float*)d_in[4];
  const float* q_b   = (const float*)d_in[5];
  const float* k_s   = (const float*)d_in[6];
  const float* k_b   = (const float*)d_in[7];

  char* ws = (char*)d_ws;
  float*          qk_raw = (float*)ws;                          // 32 MB f32
  unsigned short* q_bf   = (unsigned short*)(ws + (32u << 20)); // 8 MB bf16
  unsigned short* k_bf   = (unsigned short*)(ws + (40u << 20)); // 8 MB bf16
  unsigned short* vT     = (unsigned short*)(ws + (48u << 20)); // 8 MB bf16
  unsigned short* aout   = (unsigned short*)(ws + (56u << 20)); // 8 MB bf16

  const float inv32 = 1.0f / 32.0f;  // 1/sqrt(1024) == 1/D^C

  // qk = X @ W_qk / 32  (f32, LN comes next)
  gemm_bf16<0><<<dim3(32, 64), 128, 0, stream>>>(X, W_qk, qk_raw,
                                                 4096, 2048, 1024, inv32);
  // v = X @ W_v / 32  -> bf16, transposed to [b,h,d,l]
  gemm_bf16<1><<<dim3(16, 64), 128, 0, stream>>>(X, W_v, vT,
                                                 4096, 1024, 1024, inv32);
  // per-head layernorm of q,k -> bf16 [b,h,l,64]
  ln_qk<<<16384, 256, 0, stream>>>(qk_raw, q_s, q_b, k_s, k_b, q_bf, k_bf);
  // causal attention
  attn_kernel<<<dim3(32, 32), 128, 0, stream>>>(q_bf, k_bf, vT, aout);
  // out = attn @ W_out / 32 -> f32
  gemm_bf16<2><<<dim3(16, 64), 128, 0, stream>>>(aout, W_out, (float*)d_out,
                                                 4096, 1024, 1024, inv32);
}
